// GCN_6382321401984
// MI455X (gfx1250) — matrix operand order repaired and checked by reading, not compiled, not run on hardware
//
#include <hip/hip_runtime.h>
#include <hip/hip_bf16.h>

typedef __attribute__((ext_vector_type(16))) __bf16 v16bf;
typedef __attribute__((ext_vector_type(8)))  float  v8f;

// ---------- helpers ----------
__device__ __forceinline__ unsigned short f2bf(float f) {
  unsigned u = __float_as_uint(f);
  u += 0x7FFFu + ((u >> 16) & 1u);          // round-to-nearest-even
  return (unsigned short)(u >> 16);
}

// ---------- degree / normalization ----------
__global__ void k_deg_init(float* deg, int n) {
  int i = blockIdx.x * blockDim.x + threadIdx.x;
  if (i < n) deg[i] = 1.0f;                  // self-loop contribution
}
__global__ void k_deg_count(const int* __restrict__ dst, float* deg, int e) {
  int i = blockIdx.x * blockDim.x + threadIdx.x;
  if (i < e) atomicAdd(&deg[dst[i]], 1.0f);
}
__global__ void k_deg_rsqrt(float* deg, int n) {
  int i = blockIdx.x * blockDim.x + threadIdx.x;
  if (i < n) deg[i] = rsqrtf(deg[i]);        // deg >= 1 always
}

// ---------- weight repack: W[K][N] f32 -> Wt[N][K] bf16 ----------
__global__ void k_repack_w(const float* __restrict__ W, unsigned short* __restrict__ Wt,
                           int K, int Nout) {
  int idx = blockIdx.x * blockDim.x + threadIdx.x;
  if (idx >= K * Nout) return;
  int k = idx / Nout, nn = idx - k * Nout;
  Wt[nn * K + k] = f2bf(W[idx]);
}

// ---------- layer 1: T = x(Nx2) @ W1(2x32) ----------
__global__ void k_lin1(const float* __restrict__ x, const float* __restrict__ W1,
                       float* __restrict__ T, int n) {
  int idx = blockIdx.x * blockDim.x + threadIdx.x;
  int i = idx >> 5, j = idx & 31;
  if (i >= n) return;
  T[idx] = x[2 * i] * W1[j] + x[2 * i + 1] * W1[32 + j];
}

// ---------- WMMA GEMM: T[nRows x 128] = A[nRows x K](bf16) @ W[K x 128] ----------
// One wave per 16-row tile, computing ALL 8 column tiles (A fragment loaded once
// per K-step and reused 8x). K and Nout=128 are compile-time constants, so every
// B load and C store is an immediate offset off a single base address.
template <int K>
__global__ __launch_bounds__(256) void k_gemm_wmma(
    const unsigned short* __restrict__ Abf,
    const unsigned short* __restrict__ Wt,
    float* __restrict__ T, int fullTilesM) {
  int wave = (int)((blockIdx.x * (size_t)blockDim.x + threadIdx.x) >> 5);
  int lane = threadIdx.x & 31;
  if (wave >= fullTilesM) return;           // wave-uniform exit (EXEC all-1 for WMMA)

  int half = lane >> 4;                     // 0: lanes 0-15, 1: lanes 16-31
  int lsub = lane & 15;
  const unsigned short* arow = Abf + (size_t)(wave * 16 + lsub) * K;
  const unsigned short* wcol = Wt + (size_t)lsub * K;   // tile t adds t*16*K (immediate)

  v8f acc[8] = {};
  #pragma unroll
  for (int kk = 0; kk < K; kk += 32) {
    // A fragment (16x32 bf16): lane-half h holds K = h*16..h*16+15
    union { v16bf v; uint4 q[2]; } a;
    a.q[0] = *(const uint4*)(arow + kk + half * 16);
    a.q[1] = *(const uint4*)(arow + kk + half * 16 + 8);
    #pragma unroll
    for (int t = 0; t < 8; ++t) {
      // B fragment (32x16 bf16): lane-half h holds K = h*16..h*16+15 of one column
      union { v16bf v; uint4 q[2]; } b;
      const unsigned short* wp = wcol + (size_t)(t * 16) * K + kk + half * 16;
      b.q[0] = *(const uint4*)(wp);
      b.q[1] = *(const uint4*)(wp + 8);
      acc[t] = __builtin_amdgcn_wmma_f32_16x16x32_bf16(false, a.v, false, b.v,
                                                       (short)0, acc[t], false, false);
    }
  }

  // C layout: VGPR v, lane-half h -> row = wave*16 + h*8 + v, col = t*16 + lsub
  float* out = T + (size_t)(wave * 16 + half * 8) * 128 + lsub;
  #pragma unroll
  for (int t = 0; t < 8; ++t) {
    #pragma unroll
    for (int vv = 0; vv < 8; ++vv)
      out[t * 16 + (size_t)vv * 128] = acc[t][vv];
  }
}

// scalar fallback for remainder rows (only launched when nRows % 16 != 0)
__global__ void k_gemm_tail(const unsigned short* __restrict__ Abf,
                            const unsigned short* __restrict__ Wt,
                            float* __restrict__ T, int rowStart, int nRows, int K) {
  int idx = blockIdx.x * blockDim.x + threadIdx.x;
  int r = rowStart + (idx >> 7);
  int c = idx & 127;
  if (r >= nRows) return;
  const unsigned short* a = Abf + (size_t)r * K;
  const unsigned short* w = Wt + (size_t)c * K;
  float acc = 0.0f;
  for (int k = 0; k < K; ++k)
    acc += __uint_as_float((unsigned)a[k] << 16) * __uint_as_float((unsigned)w[k] << 16);
  T[(size_t)r * 128 + c] = acc;
}

// ---------- aggregation ----------
__global__ void k_bias_init(float* __restrict__ out, const float* __restrict__ b,
                            int total, int mask) {
  int i = blockIdx.x * blockDim.x + threadIdx.x;
  if (i < total) out[i] = b[i & mask];
}

// one thread per (edge_or_selfloop, feature); F = 1<<logF so F consecutive
// threads share an edge -> coalesced T reads, wave-uniform dis loads.
__global__ void k_scatter(const int* __restrict__ src, const int* __restrict__ dst,
                          const float* __restrict__ dis, const float* __restrict__ T,
                          float* __restrict__ out, int e, int n, int logF) {
  long long idx = (long long)blockIdx.x * blockDim.x + threadIdx.x;
  long long total = ((long long)(e + n)) << logF;
  if (idx >= total) return;
  int F  = 1 << logF;
  int ed = (int)(idx >> logF);
  int f  = (int)(idx & (F - 1));
  int s, d; float nrm;
  if (ed < e) { s = src[ed]; d = dst[ed]; nrm = dis[s] * dis[d]; }
  else        { s = d = ed - e; float v = dis[s]; nrm = v * v; }
  atomicAdd(out + (size_t)d * F + f, T[(size_t)s * F + f] * nrm);
}

// ReLU in place + emit bf16 mirror for the next layer's WMMA A-operand
__global__ void k_relu_bf(float* __restrict__ x, unsigned short* __restrict__ xbf,
                          int total) {
  int i = blockIdx.x * blockDim.x + threadIdx.x;
  if (i >= total) return;
  float v = fmaxf(x[i], 0.0f);
  x[i] = v;
  xbf[i] = f2bf(v);
}

// ---------- final 128->1 + LeakyReLU ----------
__global__ void k_final(const float* __restrict__ H, const float* __restrict__ Wl,
                        const float* __restrict__ bl, float* __restrict__ out, int n) {
  int i = blockIdx.x * blockDim.x + threadIdx.x;
  if (i >= n) return;
  const float4* h = (const float4*)(H + (size_t)i * 128);
  const float4* w = (const float4*)Wl;
  float acc = 0.0f;
  #pragma unroll
  for (int q = 0; q < 32; ++q) {
    float4 a = h[q]; float4 ww = w[q];
    acc += a.x * ww.x + a.y * ww.y + a.z * ww.z + a.w * ww.w;
  }
  acc += bl[0];
  out[i] = acc > 0.0f ? acc : 0.01f * acc;
}

// ---------- host ----------
extern "C" void kernel_launch(void* const* d_in, const int* in_sizes, int n_in,
                              void* d_out, int out_size, void* d_ws, size_t ws_size,
                              hipStream_t stream) {
  const float* x  = (const float*)d_in[0];
  const int*   ei = (const int*)d_in[1];
  const float* W1 = (const float*)d_in[2];
  const float* b1 = (const float*)d_in[3];
  const float* W2 = (const float*)d_in[4];
  const float* b2 = (const float*)d_in[5];
  const float* W3 = (const float*)d_in[6];
  const float* b3 = (const float*)d_in[7];
  const float* W4 = (const float*)d_in[8];
  const float* b4 = (const float*)d_in[9];
  const float* W5 = (const float*)d_in[10];
  const float* b5 = (const float*)d_in[11];
  const float* Wl = (const float*)d_in[12];
  const float* bl = (const float*)d_in[13];

  const int n = in_sizes[0] / 2;   // x is [N,2]
  const int e = in_sizes[1] / 2;   // edge_index is [2,E]
  const int* src = ei;
  const int* dst = ei + e;

  // workspace layout (16B aligned sections)
  float* dis  = (float*)d_ws;
  size_t np   = (size_t)((n + 3) & ~3);
  float* bufA = dis + np;                            // activations H f32 (N*128)
  float* bufB = bufA + (size_t)n * 128;              // linear output T f32 (N*128)
  unsigned short* bufAbf = (unsigned short*)(bufB + (size_t)n * 128); // H in bf16 (N*128)
  unsigned short* wt2 = bufAbf + (size_t)n * 128;
  unsigned short* wt3 = wt2 + 128 * 32;
  unsigned short* wt4 = wt3 + 128 * 128;
  unsigned short* wt5 = wt4 + 128 * 128;

  const int B = 256;
  #define GRID(t) ((int)(((long long)(t) + B - 1) / B))

  // normalization coefficients
  k_deg_init <<<GRID(n), B, 0, stream>>>(dis, n);
  k_deg_count<<<GRID(e), B, 0, stream>>>(dst, dis, e);
  k_deg_rsqrt<<<GRID(n), B, 0, stream>>>(dis, n);

  // bf16 weight repack (once)
  k_repack_w<<<GRID(32 * 128),  B, 0, stream>>>(W2, wt2, 32, 128);
  k_repack_w<<<GRID(128 * 128), B, 0, stream>>>(W3, wt3, 128, 128);
  k_repack_w<<<GRID(128 * 128), B, 0, stream>>>(W4, wt4, 128, 128);
  k_repack_w<<<GRID(128 * 128), B, 0, stream>>>(W5, wt5, 128, 128);

  // layer 1: K=2 VALU linear, then aggregate (F=32), ReLU + bf16 mirror
  k_lin1     <<<GRID((long long)n * 32), B, 0, stream>>>(x, W1, bufB, n);
  k_bias_init<<<GRID((long long)n * 32), B, 0, stream>>>(bufA, b1, n * 32, 31);
  k_scatter  <<<GRID(((long long)(e + n)) << 5), B, 0, stream>>>(src, dst, dis, bufB, bufA, e, n, 5);
  k_relu_bf  <<<GRID((long long)n * 32), B, 0, stream>>>(bufA, bufAbf, n * 32);

  // layers 2..5: WMMA linear (bf16 A, one wave per 16-row tile) + aggregate + ReLU
  const int fullTilesM = n >> 4;
  const int rem        = n & 15;
  const long long gemmThreads = (long long)fullTilesM * 32;

  // layer 2 (K = 32)
  k_gemm_wmma<32><<<GRID(gemmThreads), B, 0, stream>>>(bufAbf, wt2, bufB, fullTilesM);
  if (rem) k_gemm_tail<<<GRID(rem * 128), B, 0, stream>>>(bufAbf, wt2, bufB, fullTilesM * 16, n, 32);
  k_bias_init<<<GRID((long long)n * 128), B, 0, stream>>>(bufA, b2, n * 128, 127);
  k_scatter  <<<GRID(((long long)(e + n)) << 7), B, 0, stream>>>(src, dst, dis, bufB, bufA, e, n, 7);
  k_relu_bf  <<<GRID((long long)n * 128), B, 0, stream>>>(bufA, bufAbf, n * 128);

  // layers 3..5 (K = 128)
  const unsigned short* wts[3] = { wt3, wt4, wt5 };
  const float*          bs[3]  = { b3,  b4,  b5  };
  for (int li = 0; li < 3; ++li) {
    k_gemm_wmma<128><<<GRID(gemmThreads), B, 0, stream>>>(bufAbf, wts[li], bufB, fullTilesM);
    if (rem) k_gemm_tail<<<GRID(rem * 128), B, 0, stream>>>(bufAbf, wts[li], bufB, fullTilesM * 16, n, 128);
    k_bias_init<<<GRID((long long)n * 128), B, 0, stream>>>(bufA, bs[li], n * 128, 127);
    k_scatter  <<<GRID(((long long)(e + n)) << 7), B, 0, stream>>>(src, dst, dis, bufB, bufA, e, n, 7);
    k_relu_bf  <<<GRID((long long)n * 128), B, 0, stream>>>(bufA, bufAbf, n * 128);
  }

  // final linear + LeakyReLU
  k_final<<<GRID(n), B, 0, stream>>>(bufA, Wl, bl, (float*)d_out, n);
  #undef GRID
}
